// Retention_32186484916892
// MI455X (gfx1250) — compile-verified, matrix operand order checked by reading, not run on hardware
//
#include <hip/hip_runtime.h>
#include <hip/hip_bf16.h>

// ---------------------------------------------------------------------------
// RetNet retention, MI455X (gfx1250, wave32, WMMA bf16 + Tensor Data Mover).
//   B=8, S=2048, D=256.  out[b,n,:] = sum_{m<=n} gamma^(n-m) (qr_n . kr_m) v_m
// Kernel 1: RoPE(q,k) + cast to bf16; V is written TRANSPOSED [B,D,S].
// Kernel 2: flash-style streaming retention; KV blocks stream into
//           double-buffered LDS via TENSOR_LOAD_TO_LDS (TDM) with the D#
//           pad feature generating the bank-conflict-free strides.
// ---------------------------------------------------------------------------

typedef __bf16 bf16_t;
typedef __attribute__((ext_vector_type(16))) __bf16 v16bf;
typedef __attribute__((ext_vector_type(8)))  __bf16 v8bf;
typedef __attribute__((ext_vector_type(8)))  float  v8f;
typedef __attribute__((ext_vector_type(4)))  unsigned int v4u;
typedef __attribute__((ext_vector_type(8)))  int v8i;
typedef __attribute__((ext_vector_type(4)))  int v4i;

#define BSZ 8
#define SQL 2048
#define DH  256
#define BM  128            // query rows per workgroup (8 waves x 16)
#define BN  32             // kv columns per iteration (one A-frag K-dim)
#define K_STRIDE 264       // 256 + 8 pad  (128 DWORD rows + 4 DWORD pad)
#define V_STRIDE 40        // 32 + 8 pad   (16 DWORD rows + 4 DWORD pad)
#define P_STRIDE 40

#if __has_builtin(__builtin_amdgcn_tensor_load_to_lds)
#define HAS_TDM 1
#else
#define HAS_TDM 0
#endif

__device__ __forceinline__ bf16_t f2bf(float f) {
    union { float f; unsigned u; } a; a.f = f;
    unsigned r = a.u + 0x7FFFu + ((a.u >> 16) & 1u);   // round-to-nearest-even
    unsigned short h = (unsigned short)(r >> 16);
    union { unsigned short s; bf16_t b; } o; o.s = h;
    return o.b;
}

// Packed f32->bf16 convert + split store (rows differ, column identical):
// one v_cvt_pk_bf16_f32 feeds ds_store_b16 (lo) + ds_store_b16_d16_hi (hi).
__device__ __forceinline__ void store_pair_bf16(bf16_t* p0, bf16_t* p1, float a, float b) {
#if __has_builtin(__builtin_amdgcn_cvt_pk_bf16_f32)
    auto pk = __builtin_amdgcn_cvt_pk_bf16_f32(a, b);
    unsigned u;
    __builtin_memcpy(&u, &pk, 4);
    *(unsigned short*)p0 = (unsigned short)(u & 0xFFFFu);
    *(unsigned short*)p1 = (unsigned short)(u >> 16);
#else
    *p0 = f2bf(a);
    *p1 = f2bf(b);
#endif
}

// ISA 7.12.2 16-bit A/B fragment: per lane, two contiguous 8-elem chunks at
// row*stride + colbase + 8*half and +16 further.  p must point at chunk 0.
__device__ __forceinline__ v16bf load_frag(const bf16_t* p) {
    v8bf lo = *(const v8bf*)(p);
    v8bf hi = *(const v8bf*)(p + 16);
    v16bf out;
#pragma unroll
    for (int j = 0; j < 8; ++j) { out[j] = lo[j]; out[8 + j] = hi[j]; }
    return out;
}

// Generic pointer to LDS: the LDS offset is the low 32 bits of the address.
__device__ __forceinline__ unsigned lds_off(const void* p) {
    return (unsigned)(size_t)p;
}

#if HAS_TDM
// 2D tile DMA: global (row-major, stride0 elems/row, 2-byte elems) -> LDS,
// with pad_amount DWORDs appended every pad_interval-coded DWORDs (per row).
// 6-arg builtin form (clang-23 / therock-10.0 headers).
__device__ __forceinline__ void tdm_load_2d(const bf16_t* gsrc, unsigned ldsByteOff,
                                            unsigned tile_d0, unsigned tile_d1,
                                            unsigned tensor_d0, unsigned tensor_d1,
                                            unsigned long long stride0,
                                            unsigned pad_interval, unsigned pad_amount) {
    unsigned long long ga = (unsigned long long)(size_t)gsrc;
    v4u g0;
    g0[0] = 1u;                                              // count=1, user mode
    g0[1] = ldsByteOff;                                      // lds_addr
    g0[2] = (unsigned)(ga & 0xFFFFFFFFu);                    // global_addr[31:0]
    g0[3] = (unsigned)((ga >> 32) & 0x01FFFFFFu) | (2u << 30); // ga[56:32] | type=2
    v8i g1;
    g1[0] = (int)((1u << 16)                                 // data_size = 2 bytes
                | (1u << 20)                                 // pad_enable
                | (pad_interval << 22) | (pad_amount << 25));
    g1[1] = (int)((tensor_d0 & 0xFFFFu) << 16);              // tensor_dim0[15:0]
    g1[2] = (int)(((tensor_d0 >> 16) & 0xFFFFu) | ((tensor_d1 & 0xFFFFu) << 16));
    g1[3] = (int)(((tensor_d1 >> 16) & 0xFFFFu) | ((tile_d0 & 0xFFFFu) << 16));
    g1[4] = (int)(tile_d1 & 0xFFFFu);                        // tile_dim1, tile_dim2=0
    g1[5] = (int)(stride0 & 0xFFFFFFFFull);                  // dim0_stride[31:0]
    g1[6] = (int)((stride0 >> 32) & 0xFFFFull);              // dim0_stride[47:32]
    g1[7] = 0;
    v4i z4 = {0, 0, 0, 0};
    v8i z8 = {0, 0, 0, 0, 0, 0, 0, 0};
    __builtin_amdgcn_tensor_load_to_lds(g0, g1, z4, z4, z8, 0);
}
#endif

// ---------------------------------------------------------------------------
// Kernel 1: interleaved-pair RoPE + bf16 cast.  V written transposed [B,D,S].
// ---------------------------------------------------------------------------
__global__ void rope_cast_kernel(const float* __restrict__ q,
                                 const float* __restrict__ k,
                                 const float* __restrict__ v,
                                 bf16_t* __restrict__ qb,
                                 bf16_t* __restrict__ kb,
                                 bf16_t* __restrict__ vTb) {
    int idx = blockIdx.x * blockDim.x + threadIdx.x;   // [0, B*S*D/2) = 2^21
    int h = idx & 127;                 // pair index (D/2 = 128)
    int s = (idx >> 7) & (SQL - 1);
    int b = idx >> 18;
    float theta = exp2f(-(2.0f * (float)h / (float)DH) * 13.287712379549449f);
    float ang = (float)s * theta;
    float sn, cs;
    __sincosf(ang, &sn, &cs);
    size_t base = (size_t)idx * 2;     // == ((b*S+s)*D + 2h)

    float qe = q[base], qo = q[base + 1];
    qb[base]     = f2bf(qe * cs - qo * sn);
    qb[base + 1] = f2bf(qe * sn + qo * cs);

    float ke = k[base], ko = k[base + 1];
    kb[base]     = f2bf(ke * cs - ko * sn);
    kb[base + 1] = f2bf(ke * sn + ko * cs);

    // transposed V: vT[b][d][s]
    vTb[((size_t)b * DH + 2 * h    ) * SQL + s] = f2bf(v[base]);
    vTb[((size_t)b * DH + 2 * h + 1) * SQL + s] = f2bf(v[base + 1]);
}

// ---------------------------------------------------------------------------
// Kernel 2: streaming retention.  256 threads = 8 wave32; wave w owns rows
// [rowblk*128 + 16w, +16).  KV streamed in 32-col blocks, double-buffered,
// DMA'd by the TDM (wave 0) and overlapped with WMMA compute.
// ---------------------------------------------------------------------------
__global__ void __launch_bounds__(256, 1)
retention_kernel(const bf16_t* __restrict__ Qb, const bf16_t* __restrict__ Kb,
                 const bf16_t* __restrict__ VTb, const float* __restrict__ gamma_p,
                 float* __restrict__ out) {
    __shared__ __align__(16) bf16_t k_lds[2][BN][K_STRIDE];   // K block, row-major
    __shared__ __align__(16) bf16_t vT_lds[2][DH][V_STRIDE];  // V block, transposed
    __shared__ __align__(16) bf16_t p_lds[8][16][P_STRIDE];   // per-wave P scratch

    const int b      = blockIdx.x / (SQL / BM);
    const int rowblk = blockIdx.x % (SQL / BM);
    const int tid  = threadIdx.x;
    const int wave = tid >> 5;
    const int lane = tid & 31;
    const int r    = lane & 15;        // N / row-within-tile coordinate
    const int half = lane >> 4;        // lane group select
    const int n0w  = rowblk * BM + wave * 16;

    const float log2g = log2f(gamma_p[0]);

    // hoisted decay factors:  gamma^(n-m) = powh[i] * gamma^(n0w - m)
    float powh[8];                     // gamma^(i + 8*half)
#pragma unroll
    for (int i = 0; i < 8; ++i) powh[i] = exp2f((float)(i + 8 * half) * log2g);
    const float ginv16 = exp2f(-16.0f * log2g);   // gamma^-16

    const bf16_t* kBase  = Kb  + (size_t)b * SQL * DH;
    const bf16_t* vtBase = VTb + (size_t)b * DH * SQL;

    // Q fragments (A layout), this wave's 16 rows, K-dim 256 = 8 chunks of 32.
    v16bf qfrag[8];
    const bf16_t* qrow = Qb + (size_t)(b * SQL + n0w + r) * DH;
#pragma unroll
    for (int kc = 0; kc < 8; ++kc)
        qfrag[kc] = load_frag(qrow + kc * 32 + half * 8);

    v8f accO[16];
#pragma unroll
    for (int t = 0; t < 16; ++t) accO[t] = v8f{};

    const int nBlk = (rowblk + 1) * (BM / BN);   // causal: stream m0 < (rowblk+1)*BM

#if HAS_TDM
    // prologue: DMA block 0 into buffer 0
    if (wave == 0) {
        tdm_load_2d(kBase, lds_off(&k_lds[0][0][0]), DH, BN, DH, SQL,
                    (unsigned long long)DH, 6u, 3u);          // 128 dw rows + 4 dw pad
        tdm_load_2d(vtBase, lds_off(&vT_lds[0][0][0]), BN, DH, SQL, DH,
                    (unsigned long long)SQL, 3u, 3u);         // 16 dw rows + 4 dw pad
    }
#endif

    for (int blk = 0; blk < nBlk; ++blk) {
        const int m0  = blk * BN;
        const int cur = blk & 1;

#if HAS_TDM
        __syncthreads();      // everyone done reading buffer cur^1 (block blk-1)
        if (wave == 0) {
            if (blk + 1 < nBlk) {
                const int nm0 = m0 + BN;
                tdm_load_2d(kBase + (size_t)nm0 * DH, lds_off(&k_lds[cur ^ 1][0][0]),
                            DH, BN, DH, SQL, (unsigned long long)DH, 6u, 3u);
                tdm_load_2d(vtBase + nm0, lds_off(&vT_lds[cur ^ 1][0][0]),
                            BN, DH, SQL, DH, (unsigned long long)SQL, 3u, 3u);
                __builtin_amdgcn_s_wait_tensorcnt((short)2);  // block blk complete
            } else {
                __builtin_amdgcn_s_wait_tensorcnt((short)0);
            }
        }
        __syncthreads();      // buffer cur published to all waves
#else
        __syncthreads();
        {   // cooperative fallback load
            const bf16_t* ksrc = kBase + (size_t)m0 * DH;
#pragma unroll
            for (int it = 0; it < 4; ++it) {
                int c = tid + it * 256, row = c >> 5, cc = (c & 31) * 8;
                *(uint4*)&k_lds[cur][row][cc] = *(const uint4*)(ksrc + (size_t)row * DH + cc);
            }
            const bf16_t* vsrc = vtBase + m0;   // vT rows: stride SQL
#pragma unroll
            for (int c = 0; c < 4; ++c)
                *(uint4*)&vT_lds[cur][tid][c * 8] = *(const uint4*)(vsrc + (size_t)tid * SQL + c * 8);
        }
        __syncthreads();
#endif

        if (m0 <= n0w + 15) {     // wave-uniform: EXEC stays all-ones in WMMA
            const bf16_t (*K)[K_STRIDE] = k_lds[cur];
            const bf16_t (*VT)[V_STRIDE] = vT_lds[cur];

            // ---- S = Q . K^T : two 16x16 col tiles, K-dim 256, pipelined ----
            v8f s0 = v8f{}, s1 = v8f{};
            v16bf fb0 = load_frag(&K[r][half * 8]);
            v16bf fb1 = load_frag(&K[16 + r][half * 8]);
#pragma unroll
            for (int kc = 0; kc < 8; ++kc) {
                v16bf nb0 = fb0, nb1 = fb1;
                if (kc < 7) {
                    nb0 = load_frag(&K[r][(kc + 1) * 32 + half * 8]);
                    nb1 = load_frag(&K[16 + r][(kc + 1) * 32 + half * 8]);
                }
                s0 = __builtin_amdgcn_wmma_f32_16x16x32_bf16(false, qfrag[kc], false, fb0,
                                                             (short)0, s0, false, false);
                s1 = __builtin_amdgcn_wmma_f32_16x16x32_bf16(false, qfrag[kc], false, fb1,
                                                             (short)0, s1, false, false);
                fb0 = nb0; fb1 = nb1;
            }

            // ---- causal decay gamma^(n-m); C layout: M = i+8*half, N = r ----
            bf16_t (*P)[P_STRIDE] = p_lds[wave];
            if (m0 + BN - 1 <= n0w) {
                // block fully below diagonal: no mask, factored decay
                float base0 = exp2f((float)(n0w - m0 - r) * log2g);  // col tile 0
                float base1 = base0 * ginv16;                        // col tile 1
#pragma unroll
                for (int i = 0; i < 8; i += 2) {
                    store_pair_bf16(&P[i + 8 * half][r], &P[i + 1 + 8 * half][r],
                                    s0[i] * powh[i] * base0, s0[i + 1] * powh[i + 1] * base0);
                    store_pair_bf16(&P[i + 8 * half][r + 16], &P[i + 1 + 8 * half][r + 16],
                                    s1[i] * powh[i] * base1, s1[i + 1] * powh[i + 1] * base1);
                }
            } else {
                // diagonal block: per-element causal mask
#pragma unroll
                for (int i = 0; i < 8; i += 2) {
                    float d0a, d0b, d1a, d1b;
                    {
                        int na = n0w + i + 8 * half, nb2 = na + 1;
                        int mc0 = m0 + r, mc1 = m0 + 16 + r;
                        d0a = (na  >= mc0) ? exp2f((float)(na  - mc0) * log2g) : 0.f;
                        d0b = (nb2 >= mc0) ? exp2f((float)(nb2 - mc0) * log2g) : 0.f;
                        d1a = (na  >= mc1) ? exp2f((float)(na  - mc1) * log2g) : 0.f;
                        d1b = (nb2 >= mc1) ? exp2f((float)(nb2 - mc1) * log2g) : 0.f;
                    }
                    store_pair_bf16(&P[i + 8 * half][r], &P[i + 1 + 8 * half][r],
                                    s0[i] * d0a, s0[i + 1] * d0b);
                    store_pair_bf16(&P[i + 8 * half][r + 16], &P[i + 1 + 8 * half][r + 16],
                                    s1[i] * d1a, s1[i + 1] * d1b);
                }
            }

            // per-wave LDS round-trip; DS pipe is in-order within a wave
            v16bf pfrag = load_frag(&P[r][half * 8]);
            // ---- O += P . V : 16 output col tiles, pipelined ----
            v16bf vf = load_frag(&VT[r][half * 8]);
#pragma unroll
            for (int t = 0; t < 16; ++t) {
                v16bf nvf = vf;
                if (t < 15) nvf = load_frag(&VT[(t + 1) * 16 + r][half * 8]);
                accO[t] = __builtin_amdgcn_wmma_f32_16x16x32_bf16(false, pfrag, false, vf,
                                                                  (short)0, accO[t], false, false);
                vf = nvf;
            }
        }
    }

    // ---- writeback: C layout -> out[b, n0w + i + 8*half, 16t + r] ----
    float* orow = out + (size_t)(b * SQL + n0w) * DH;
#pragma unroll
    for (int t = 0; t < 16; ++t)
#pragma unroll
        for (int i = 0; i < 8; ++i)
            orow[(size_t)(i + 8 * half) * DH + t * 16 + r] = accO[t][i];
}

// ---------------------------------------------------------------------------
extern "C" void kernel_launch(void* const* d_in, const int* in_sizes, int n_in,
                              void* d_out, int out_size, void* d_ws, size_t ws_size,
                              hipStream_t stream) {
    const float* q = (const float*)d_in[0];
    const float* k = (const float*)d_in[1];
    const float* v = (const float*)d_in[2];
    const float* gamma = (const float*)d_in[3];
    float* out = (float*)d_out;

    const size_t nElem = (size_t)BSZ * SQL * DH;       // 4,194,304
    bf16_t* qb  = (bf16_t*)d_ws;                       // 8 MB
    bf16_t* kb  = qb + nElem;                          // 8 MB
    bf16_t* vtb = kb + nElem;                          // 8 MB (24 MB total)

    rope_cast_kernel<<<(int)(nElem / 2 / 256), 256, 0, stream>>>(q, k, v, qb, kb, vtb);
    retention_kernel<<<BSZ * (SQL / BM), 256, 0, stream>>>(qb, kb, vtb, gamma, out);
}
